// GCViTBlock_86904368267309
// MI455X (gfx1250) — matrix operand order repaired and checked by reading, not compiled, not run on hardware
//
#include <hip/hip_runtime.h>
#include <math.h>
#include <stdint.h>

#define DI __device__ __forceinline__

typedef __attribute__((ext_vector_type(16))) __bf16 v16bf;
typedef __attribute__((ext_vector_type(8)))  __bf16 v8bf;
typedef __attribute__((ext_vector_type(8)))  float  v8f;
typedef __attribute__((ext_vector_type(4)))  unsigned int u32x4;
typedef __attribute__((ext_vector_type(8)))  int i32x8;
typedef __attribute__((ext_vector_type(4)))  int i32x4;

// ---- problem constants (H=8, W=256 path: wh=8, ww=16) ----
constexpr int C_     = 768;
constexpr int NH_    = 16;
constexpr int HD_    = 48;
constexpr int KPAD   = 64;       // HD padded to 64 for K=32 WMMA steps
constexpr int N_     = 128;      // tokens per window
constexpr int BW_    = 512;      // B * nwin = 32*16
constexpr int MTOK   = BW_ * N_; // 65536 rows
constexpr int HID_   = 3072;

// ---- CDNA5 TDM availability (device pass on ROCm>=7.2 has the builtin) ----
#if defined(__has_builtin)
#if __has_builtin(__builtin_amdgcn_tensor_load_to_lds)
#define CDNA5_TDM 1
#endif
#endif
#ifndef CDNA5_TDM
#define CDNA5_TDM 0
#endif

DI unsigned short f2bf(float f) {
  unsigned u = __float_as_uint(f);
  unsigned r = (u + 0x7fffu + ((u >> 16) & 1u)) >> 16;
  return (unsigned short)r;
}

// A-fragment: 16x32 bf16, lane half h -> contiguous K [16h..16h+15]
DI v16bf fragA(const unsigned short* rowp, int h) {
  v8bf lo = *reinterpret_cast<const v8bf*>(rowp + h * 16);
  v8bf hi = *reinterpret_cast<const v8bf*>(rowp + h * 16 + 8);
  return __builtin_shufflevector(lo, hi, 0,1,2,3,4,5,6,7,8,9,10,11,12,13,14,15);
}
// B-fragment: 32x16 bf16, lane holds one column: 16 contiguous K halfs of B^T row
DI v16bf fragB(const unsigned short* p) {
  v8bf lo = *reinterpret_cast<const v8bf*>(p);
  v8bf hi = *reinterpret_cast<const v8bf*>(p + 8);
  return __builtin_shufflevector(lo, hi, 0,1,2,3,4,5,6,7,8,9,10,11,12,13,14,15);
}
DI v8f wmma_bf16(v16bf a, v16bf b, v8f c) {
  return __builtin_amdgcn_wmma_f32_16x16x32_bf16(false, a, false, b, (short)0, c, false, false);
}
DI void zero_acc(v8f& a) {
  #pragma unroll
  for (int j = 0; j < 8; j++) a[j] = 0.f;
}

DI unsigned lds_addr_of(const void* p) {   // flat LDS addr truncates to LDS offset (ISA 10.2)
  return (unsigned)(unsigned long long)(uintptr_t)p;
}
DI void wait_tensor0() {
#if defined(__has_builtin) && __has_builtin(__builtin_amdgcn_s_wait_tensorcnt)
  __builtin_amdgcn_s_wait_tensorcnt(0);
#else
  asm volatile("s_wait_tensorcnt 0x0" ::: "memory");
#endif
}

// Issue one TDM 2D tile load: 128 rows x 64 halfs (bf16), row stride = strideHalves,
// LDS destination padded 128B data + 16B pad -> 144B (72-half) row stride.
DI void tdm_issue_tile(const unsigned short* gptr, int strideHalves, unsigned ldsByteAddr) {
#if CDNA5_TDM
  unsigned long long ga = (unsigned long long)(uintptr_t)gptr;
  u32x4 g0;
  g0[0] = 1u;                                   // count=1, user mode, no gather
  g0[1] = ldsByteAddr;                          // lds_addr
  g0[2] = (unsigned)ga;                         // global_addr[31:0]
  g0[3] = (unsigned)(ga >> 32) | 0x80000000u;   // global_addr[56:32], type=2 ("image")
  i32x8 g1;
  g1[0] = 0x07110000;    // data_size=2B, pad_enable, pad_interval=32DW(128B), pad_amount=4DW(16B)
  g1[1] = (int)(64u << 16);    // tensor_dim0 = 64 (low16 at bit48)
  g1[2] = (int)(128u << 16);   // tensor_dim0 hi=0 | tensor_dim1 = 128 (low16 at bit80)
  g1[3] = (int)(64u << 16);    // tensor_dim1 hi=0 | tile_dim0 = 64
  g1[4] = 128;                 // tile_dim1 = 128, tile_dim2 = 0
  g1[5] = strideHalves;        // tensor_dim0_stride low32
  g1[6] = 0;                   // stride0 hi | stride1 lo
  g1[7] = 0;
  i32x4 z4; z4[0] = z4[1] = z4[2] = z4[3] = 0;
#if __clang_major__ >= 23
  i32x8 z8; z8[0]=z8[1]=z8[2]=z8[3]=z8[4]=z8[5]=z8[6]=z8[7]=0;
  __builtin_amdgcn_tensor_load_to_lds(g0, g1, z4, z4, z8, 0);
#else
  __builtin_amdgcn_tensor_load_to_lds(g0, g1, z4, z4, 0);
#endif
#else
  (void)gptr; (void)strideHalves; (void)ldsByteAddr;
#endif
}

// ---------------- LayerNorm (one wave per token, 768 channels) ----------------
__global__ __launch_bounds__(256) void ln_kernel(
    const float* __restrict__ in, const float* __restrict__ gw,
    const float* __restrict__ bw_, float* __restrict__ shortcut,
    unsigned short* __restrict__ out_bf, int mode) {
  const int wid = threadIdx.x >> 5, lane = threadIdx.x & 31;
  const int t = blockIdx.x * 8 + wid;  // token id, < 65536
  size_t dstrow;
  if (mode == 0) {
    int b = t >> 11, pos = t & 2047;
    int hr = pos >> 8, wc = pos & 255;
    int wb = wc >> 4, wi = wc & 15;
    dstrow = (size_t)((b * 16 + wb) * 128 + hr * 16 + wi);
  } else {
    dstrow = (size_t)t;
  }
  const size_t src = (size_t)t * C_;
  float xv[24], s = 0.f, s2 = 0.f;
  #pragma unroll
  for (int i = 0; i < 24; i++) {
    float v = in[src + i * 32 + lane];
    xv[i] = v; s += v; s2 += v * v;
  }
  #pragma unroll
  for (int o = 16; o > 0; o >>= 1) { s += __shfl_xor(s, o, 32); s2 += __shfl_xor(s2, o, 32); }
  const float mean = s * (1.f / 768.f);
  const float var  = s2 * (1.f / 768.f) - mean * mean;
  const float rstd = rsqrtf(var + 1e-5f);
  const size_t dbase = dstrow * C_;
  #pragma unroll
  for (int i = 0; i < 24; i++) {
    int c = i * 32 + lane;
    float y = (xv[i] - mean) * rstd * gw[c] + bw_[c];
    if (mode == 0) shortcut[dbase + c] = y;
    out_bf[dbase + c] = f2bf(y);
  }
}

// ---------------- small prep kernels ----------------
__global__ void convert_bf16(const float* __restrict__ in, unsigned short* __restrict__ out, int n) {
  for (int i = blockIdx.x * blockDim.x + threadIdx.x; i < n; i += gridDim.x * blockDim.x)
    out[i] = f2bf(in[i]);
}

__global__ void q_prep(const float* __restrict__ q, unsigned short* __restrict__ qb) {
  int i = blockIdx.x * blockDim.x + threadIdx.x;
  if (i >= 32 * 16 * 128 * 48) return;
  int d = i % 48;
  int n = (i / 48) & 127;
  int h = (i / (48 * 128)) & 15;
  int b = i / (48 * 128 * 16);
  const float scale = 0.14433756729740643f;  // 48^-0.5
  qb[(size_t)((b * 16 + h) * 128 + n) * KPAD + d] = f2bf(q[i] * scale);
}

__global__ void bias_prep(const float* __restrict__ rpb, float* __restrict__ bt) {
  int i = blockIdx.x * blockDim.x + threadIdx.x;
  if (i >= 128 * 128) return;
  int n = i >> 7, m = i & 127;
  int hi_ = n >> 4, wi_ = n & 15, hj = m >> 4, wj = m & 15;
  int ridx = (hi_ - hj + 7) * 63 + (wi_ - wj + 31);
  #pragma unroll
  for (int h = 0; h < 16; h++)
    bt[(size_t)((h << 7) + n) * 128 + m] = rpb[ridx * 16 + h];
}

// ---------------- generic bf16 WMMA GEMM: out = A[M,K] @ W[Nout,K]^T + bias ----------------
// 128x128 macro-tile, 64-K slabs, double-buffered LDS (stride 72 halfs = 144B, bank-spread),
// staged by the Tensor Data Mover when available (one issuing wave, TENSORcnt-tracked).
constexpr int MODE_KV = 0, MODE_PROJ = 1, MODE_FC1 = 2, MODE_FC2 = 3;
constexpr int LDS_STRIDE = 72;                 // 64 data halfs + 8 pad halfs
constexpr int BUFH = 128 * LDS_STRIDE;         // halfs per tile buffer
constexpr int GEMM_SMEM = 4 * BUFH * 2;        // A(2) + B(2) buffers, bytes

template <int MODE>
__global__ __launch_bounds__(256) void gemm_bf16(
    const unsigned short* __restrict__ A, const unsigned short* __restrict__ Wt,
    const float* __restrict__ bias, int K,
    unsigned short* __restrict__ kv_k, unsigned short* __restrict__ kv_vT,
    float* __restrict__ xa, unsigned short* __restrict__ out_bf,
    float* __restrict__ out_f) {
  extern __shared__ unsigned short smem[];
  unsigned short* As = smem;              // [2][128*72]
  unsigned short* Bs = smem + 2 * BUFH;   // [2][128*72]
  const int tid = threadIdx.x;
  const int w = tid >> 5, lane = tid & 31;
  const int lh = lane >> 4, lm = lane & 15;
  const size_t mbase = (size_t)blockIdx.y * 128;
  const int nbase = blockIdx.x * 128;
  v8f acc[8];
  #pragma unroll
  for (int i = 0; i < 8; i++) zero_acc(acc[i]);

  const int nslab = K >> 6;
  const unsigned short* Agp = A + mbase * (size_t)K;
  const unsigned short* Bgp = Wt + (size_t)nbase * K;

#if CDNA5_TDM
  if (tid < 32) {                       // one wave drives the TDM
    tdm_issue_tile(Agp, K, lds_addr_of(As));
    tdm_issue_tile(Bgp, K, lds_addr_of(Bs));
    wait_tensor0();
  }
  __syncthreads();
#else
  const int ldrow = tid >> 1;
  const int ldoff = (tid & 1) * 32;     // 32 halfs = 4 x uint4 per thread per tile
  uint4 ra[4], rb[4];
  {
    const uint4* ga = reinterpret_cast<const uint4*>(Agp + (size_t)ldrow * K + ldoff);
    const uint4* gb = reinterpret_cast<const uint4*>(Bgp + (size_t)ldrow * K + ldoff);
    #pragma unroll
    for (int i = 0; i < 4; i++) { ra[i] = ga[i]; rb[i] = gb[i]; }
  }
#endif

  for (int s = 0; s < nslab; s++) {
    const int buf = s & 1;
    const unsigned short* Ab = As + buf * BUFH;
    const unsigned short* Bb = Bs + buf * BUFH;
#if CDNA5_TDM
    if (tid < 32 && s + 1 < nslab) {    // prefetch next slab into the other buffer
      tdm_issue_tile(Agp + (size_t)(s + 1) * 64, K, lds_addr_of(As + (buf ^ 1) * BUFH));
      tdm_issue_tile(Bgp + (size_t)(s + 1) * 64, K, lds_addr_of(Bs + (buf ^ 1) * BUFH));
    }
#else
    {
      uint4* pa = reinterpret_cast<uint4*>(As + buf * BUFH + ldrow * LDS_STRIDE + ldoff);
      uint4* pb = reinterpret_cast<uint4*>(Bs + buf * BUFH + ldrow * LDS_STRIDE + ldoff);
      #pragma unroll
      for (int i = 0; i < 4; i++) { pa[i] = ra[i]; pb[i] = rb[i]; }
    }
    __syncthreads();
    if (s + 1 < nslab) {
      const uint4* ga = reinterpret_cast<const uint4*>(Agp + (size_t)ldrow * K + (s + 1) * 64 + ldoff);
      const uint4* gb = reinterpret_cast<const uint4*>(Bgp + (size_t)ldrow * K + (s + 1) * 64 + ldoff);
      #pragma unroll
      for (int i = 0; i < 4; i++) { ra[i] = ga[i]; rb[i] = gb[i]; }
    }
#endif
    // compute: 2 K-steps of 32; preload all 8 B-fragments so WMMAs issue back-to-back
    #pragma unroll
    for (int ks = 0; ks < 64; ks += 32) {
      v16bf bfr[8];
      #pragma unroll
      for (int nt = 0; nt < 8; nt++)
        bfr[nt] = fragB(Bb + (nt * 16 + lm) * LDS_STRIDE + ks + lh * 16);
      v16bf af = fragA(Ab + (w * 16 + lm) * LDS_STRIDE + ks, lh);
      #pragma unroll
      for (int nt = 0; nt < 8; nt++) acc[nt] = wmma_bf16(af, bfr[nt], acc[nt]);
    }
    __syncthreads();
#if CDNA5_TDM
    if (s + 1 < nslab) {
      if (tid < 32) wait_tensor0();
      __syncthreads();
    }
#endif
  }

  // fused epilogues
  #pragma unroll
  for (int nt = 0; nt < 8; nt++) {
    const int col = nbase + nt * 16 + lm;
    const float bv = bias[col];
    #pragma unroll
    for (int r = 0; r < 8; r++) {
      const size_t gm = mbase + w * 16 + lh * 8 + r;
      float val = acc[nt][r] + bv;
      if constexpr (MODE == MODE_KV) {
        int kv2 = col / 768, rem = col % 768;
        int h = rem / 48, d = rem % 48;
        int bw = (int)(gm >> 7), n = (int)(gm & 127);
        if (kv2 == 0)
          kv_k[(size_t)((bw * 16 + h) * 128 + n) * KPAD + d] = f2bf(val);
        else
          kv_vT[((size_t)(bw * 16 + h) * 48 + d) * 128 + n] = f2bf(val);
      } else if constexpr (MODE == MODE_PROJ) {
        size_t idx = gm * 768 + col;            // xa holds shortcut; in-place residual
        xa[idx] = xa[idx] + val;
      } else if constexpr (MODE == MODE_FC1) {
        float g = 0.5f * val * (1.0f + erff(val * 0.70710678118654752f));  // exact GELU
        out_bf[gm * (size_t)HID_ + col] = f2bf(g);
      } else {  // MODE_FC2: residual + window reverse scatter
        size_t idx = gm * 768 + col;
        float y = xa[idx] + val;
        int n = (int)(gm & 127), bw = (int)(gm >> 7);
        int b = bw >> 4, wb = bw & 15, hr = n >> 4, wi = n & 15;
        out_f[((size_t)b * 2048 + hr * 256 + wb * 16 + wi) * 768 + col] = y;
      }
    }
  }
}

// ---------------- fused windowed attention: one block per (window, head) ----------------
__global__ __launch_bounds__(256) void attn_kernel(
    const unsigned short* __restrict__ qb, const unsigned short* __restrict__ kb,
    const unsigned short* __restrict__ vT, const float* __restrict__ biasTab,
    unsigned short* __restrict__ attn_out) {
  extern __shared__ float S[];  // S: 128x132 f32, then P: 128x136 bf16
  unsigned short* P = reinterpret_cast<unsigned short*>(S + 128 * 132);
  const int tid = threadIdx.x, w = tid >> 5, lane = tid & 31;
  const int lh = lane >> 4, lm = lane & 15;
  const int bw = blockIdx.x >> 4, h = blockIdx.x & 15, b = bw >> 4;
  const unsigned short* qp = qb + (size_t)(b * 16 + h) * 128 * KPAD;
  const unsigned short* kp = kb + (size_t)(bw * 16 + h) * 128 * KPAD;
  const unsigned short* vp = vT + (size_t)(bw * 16 + h) * 48 * 128;

  // S = (q * hd^-0.5) @ k^T   (K padded 48->64, 2 WMMA steps, B-frags preloaded)
  v8f acc[8];
  #pragma unroll
  for (int i = 0; i < 8; i++) zero_acc(acc[i]);
  #pragma unroll
  for (int ks = 0; ks < KPAD; ks += 32) {
    v16bf bfr[8];
    #pragma unroll
    for (int nt = 0; nt < 8; nt++)
      bfr[nt] = fragB(kp + (nt * 16 + lm) * KPAD + ks + lh * 16);
    v16bf af = fragA(qp + (w * 16 + lm) * KPAD + ks, lh);
    #pragma unroll
    for (int nt = 0; nt < 8; nt++) acc[nt] = wmma_bf16(af, bfr[nt], acc[nt]);
  }
  #pragma unroll
  for (int nt = 0; nt < 8; nt++)
    #pragma unroll
    for (int r = 0; r < 8; r++)
      S[(w * 16 + lh * 8 + r) * 132 + nt * 16 + lm] = acc[nt][r];
  __syncthreads();

  // softmax over m with relative-position bias, f32, one thread per row
  if (tid < 128) {
    const int n = tid;
    const float* br = biasTab + ((size_t)(h * 128 + n)) * 128;
    float* sr = &S[n * 132];
    float mx = -1e30f;
    for (int m = 0; m < 128; m++) { float v = sr[m] + br[m]; sr[m] = v; mx = fmaxf(mx, v); }
    float sum = 0.f;
    for (int m = 0; m < 128; m++) { float e = __expf(sr[m] - mx); sr[m] = e; sum += e; }
    const float inv = 1.f / sum;
    unsigned short* pr = &P[n * 136];
    for (int m = 0; m < 128; m++) pr[m] = f2bf(sr[m] * inv);
  }
  __syncthreads();

  // out = P @ v  (v stored transposed [48][128], 4 K-steps of 32, B-frags preloaded)
  v8f acc2[3];
  #pragma unroll
  for (int i = 0; i < 3; i++) zero_acc(acc2[i]);
  #pragma unroll
  for (int kst = 0; kst < 4; kst++) {
    v16bf bfr[3];
    #pragma unroll
    for (int dt = 0; dt < 3; dt++)
      bfr[dt] = fragB(vp + (dt * 16 + lm) * 128 + kst * 32 + lh * 16);
    v16bf af = fragA(&P[(w * 16 + lm) * 136 + kst * 32], lh);
    #pragma unroll
    for (int dt = 0; dt < 3; dt++) acc2[dt] = wmma_bf16(af, bfr[dt], acc2[dt]);
  }
  #pragma unroll
  for (int dt = 0; dt < 3; dt++)
    #pragma unroll
    for (int r = 0; r < 8; r++) {
      int n = w * 16 + lh * 8 + r;
      int col = h * 48 + dt * 16 + lm;  // heads concatenated -> [Bw*N, C]
      attn_out[((size_t)bw * 128 + n) * 768 + col] = f2bf(acc2[dt][r]);
    }
}

extern "C" void kernel_launch(void* const* d_in, const int* in_sizes, int n_in,
                              void* d_out, int out_size, void* d_ws, size_t ws_size,
                              hipStream_t stream) {
  (void)in_sizes; (void)n_in; (void)out_size; (void)ws_size;
  const float* x     = (const float*)d_in[0];
  const float* qg    = (const float*)d_in[1];
  const float* n1g   = (const float*)d_in[2];
  const float* n1b   = (const float*)d_in[3];
  const float* qkvw  = (const float*)d_in[4];
  const float* qkvb  = (const float*)d_in[5];
  const float* rpb   = (const float*)d_in[6];
  const float* projw = (const float*)d_in[7];
  const float* projb = (const float*)d_in[8];
  const float* n2g   = (const float*)d_in[9];
  const float* n2b   = (const float*)d_in[10];
  const float* fc1w  = (const float*)d_in[11];
  const float* fc1b  = (const float*)d_in[12];
  const float* fc2w  = (const float*)d_in[13];
  const float* fc2b  = (const float*)d_in[14];
  float* out = (float*)d_out;

  char* ws = (char*)d_ws;
  size_t off = 0;
  auto alloc = [&](size_t bytes) -> void* {
    void* p = ws + off;
    off += (bytes + 255) & ~(size_t)255;
    return p;
  };
  float*          xa    = (float*)alloc((size_t)MTOK * C_ * 4);          // shortcut, then xa (in-place)
  unsigned short* xw    = (unsigned short*)alloc((size_t)MTOK * C_ * 2); // LN1 out -> attn out -> LN2 out
  unsigned short* wqkv  = (unsigned short*)alloc((size_t)2 * C_ * C_ * 2);
  unsigned short* wproj = (unsigned short*)alloc((size_t)C_ * C_ * 2);
  unsigned short* wfc1  = (unsigned short*)alloc((size_t)HID_ * C_ * 2);
  unsigned short* wfc2  = (unsigned short*)alloc((size_t)C_ * HID_ * 2);
  size_t qb_bytes = (size_t)32 * NH_ * N_ * KPAD * 2;
  size_t kb_bytes = (size_t)BW_ * NH_ * N_ * KPAD * 2;
  unsigned short* qbuf  = (unsigned short*)alloc(qb_bytes);
  unsigned short* kbuf  = (unsigned short*)alloc(kb_bytes);
  unsigned short* vtbuf = (unsigned short*)alloc((size_t)BW_ * NH_ * HD_ * N_ * 2);
  float*          btab  = (float*)alloc((size_t)NH_ * N_ * N_ * 4);
  unsigned short* gbuf  = (unsigned short*)alloc((size_t)MTOK * HID_ * 2);

  // zero the K-pad regions of q/k (HD 48 -> 64)
  hipMemsetAsync(qbuf, 0, qb_bytes, stream);
  hipMemsetAsync(kbuf, 0, kb_bytes, stream);

  // weight conversion + q scaling + bias table
  convert_bf16<<<2048, 256, 0, stream>>>(qkvw,  wqkv,  2 * C_ * C_);
  convert_bf16<<<2048, 256, 0, stream>>>(projw, wproj, C_ * C_);
  convert_bf16<<<2048, 256, 0, stream>>>(fc1w,  wfc1,  HID_ * C_);
  convert_bf16<<<2048, 256, 0, stream>>>(fc2w,  wfc2,  C_ * HID_);
  q_prep<<<(32 * 16 * 128 * 48 + 255) / 256, 256, 0, stream>>>(qg, qbuf);
  bias_prep<<<64, 256, 0, stream>>>(rpb, btab);

  // raise dynamic-LDS caps (GEMM 72KB, attention 100KB; WGP has 320KB)
  (void)hipFuncSetAttribute((const void*)gemm_bf16<MODE_KV>,
                            hipFuncAttributeMaxDynamicSharedMemorySize, GEMM_SMEM);
  (void)hipFuncSetAttribute((const void*)gemm_bf16<MODE_PROJ>,
                            hipFuncAttributeMaxDynamicSharedMemorySize, GEMM_SMEM);
  (void)hipFuncSetAttribute((const void*)gemm_bf16<MODE_FC1>,
                            hipFuncAttributeMaxDynamicSharedMemorySize, GEMM_SMEM);
  (void)hipFuncSetAttribute((const void*)gemm_bf16<MODE_FC2>,
                            hipFuncAttributeMaxDynamicSharedMemorySize, GEMM_SMEM);
  constexpr int ATTN_SMEM = 128 * 132 * 4 + 128 * 136 * 2;
  (void)hipFuncSetAttribute((const void*)attn_kernel,
                            hipFuncAttributeMaxDynamicSharedMemorySize, ATTN_SMEM);

  // LN1 + window partition
  ln_kernel<<<MTOK / 8, 256, 0, stream>>>(x, n1g, n1b, xa, xw, 0);

  // KV projection (scatter k row-major padded, v transposed)
  gemm_bf16<MODE_KV><<<dim3(12, MTOK / 128), 256, GEMM_SMEM, stream>>>(
      xw, wqkv, qkvb, C_, kbuf, vtbuf, nullptr, nullptr, nullptr);

  // fused attention
  attn_kernel<<<BW_ * NH_, 256, ATTN_SMEM, stream>>>(qbuf, kbuf, vtbuf, btab, xw);

  // proj + residual (xa += proj(attn))
  gemm_bf16<MODE_PROJ><<<dim3(6, MTOK / 128), 256, GEMM_SMEM, stream>>>(
      xw, wproj, projb, C_, nullptr, nullptr, xa, nullptr, nullptr);

  // LN2
  ln_kernel<<<MTOK / 8, 256, 0, stream>>>(xa, n2g, n2b, nullptr, xw, 1);

  // fc1 + exact GELU
  gemm_bf16<MODE_FC1><<<dim3(24, MTOK / 128), 256, GEMM_SMEM, stream>>>(
      xw, wfc1, fc1b, C_, nullptr, nullptr, nullptr, gbuf, nullptr);

  // fc2 + residual + window reverse -> final output
  gemm_bf16<MODE_FC2><<<dim3(6, MTOK / 128), 256, GEMM_SMEM, stream>>>(
      gbuf, wfc2, fc2b, HID_, nullptr, nullptr, xa, nullptr, out);
}
